// RNNLM_30571577213273
// MI455X (gfx1250) — compile-verified, matrix operand order checked by reading, not run on hardware
//
#include <hip/hip_runtime.h>

// RNN LM: VOCAB=32000, E=H=100, B=16, S=256
// d_in: x(int[16,256]), hidden(f32[16,100]), emb(f32[32000,100]), W_ih(f32[100,100]),
//       W_hh(f32[100,100]), b_ih(f32[100]), b_hh(f32[100]), W_fc(f32[100,32000]), b_fc(f32[32000])
// d_out: log_probs f32[16,256,32000] ++ h_last f32[16,100]
// Workspace layout (needs ~10.9 MB):
//   pre  f32[S][B][H]        @ 0         (1,638,400 B)
//   hs16 f16[4096][128]      @ 1,638,400 (1,048,576 B)   rows r=b*S+t, K padded to 128
//   wT   f16[32000][128]     @ 2,686,976 (8,192,000 B)   W_fc transposed, K padded

typedef float     v2f  __attribute__((ext_vector_type(2)));
typedef float     v8f  __attribute__((ext_vector_type(8)));
typedef _Float16  v8h  __attribute__((ext_vector_type(8)));
typedef _Float16  v16h __attribute__((ext_vector_type(16)));

#define VOCAB 32000
#define HID   100
#define BATCH 16
#define SEQ   256
#define KPAD  128
#define NROWS (BATCH*SEQ)          // 4096
#define WT_STRIDE 102              // LDS stride for W_hh^T (even, breaks 64-bank conflicts)
#define H_STRIDE  106              // LDS stride for h (even, breaks 64-bank conflicts)

// ---------------------------------------------------------------------------
// Kernel A: pre[t][b][h] = emb[x[b,t]] . W_ih[:,h] + b_ih[h] + b_hh[h]
// ---------------------------------------------------------------------------
__global__ __launch_bounds__(256) void pre_kernel(
    const int* __restrict__ x, const float* __restrict__ emb,
    const float* __restrict__ W_ih, const float* __restrict__ b_ih,
    const float* __restrict__ b_hh, float* __restrict__ pre) {
  int idx = blockIdx.x * 256 + threadIdx.x;
  if (idx >= SEQ * BATCH * HID) return;
  int t = idx / (BATCH * HID);
  int rem = idx - t * (BATCH * HID);
  int b = rem / HID;
  int h = rem - b * HID;
  int tok = x[b * SEQ + t];
  const float* e = emb + (size_t)tok * HID;
  float acc = b_ih[h] + b_hh[h];
#pragma unroll 4
  for (int k = 0; k < HID; ++k) acc = fmaf(e[k], W_ih[k * HID + h], acc);
  pre[idx] = acc;
}

// ---------------------------------------------------------------------------
// Transpose+convert W_fc [100][32000] f32 -> wT [32000][128] f16 (zero pad K)
// grid (500, 8): 64 vocab cols x 16 k rows per block
// ---------------------------------------------------------------------------
__global__ __launch_bounds__(256) void transpose_wfc(
    const float* __restrict__ W_fc, _Float16* __restrict__ wT) {
  __shared__ float tile[16][65];
  int v0 = blockIdx.x * 64;
  int k0 = blockIdx.y * 16;
  int tid = threadIdx.x;
  for (int i = tid; i < 16 * 64; i += 256) {          // coalesced read along v
    int k = i >> 6, v = i & 63;
    int gk = k0 + k;
    tile[k][v] = (gk < HID) ? W_fc[(size_t)gk * VOCAB + v0 + v] : 0.0f;
  }
  __syncthreads();
  for (int i = tid; i < 16 * 64; i += 256) {          // contiguous 32B per row write
    int v = i >> 4, k = i & 15;
    wT[(size_t)(v0 + v) * KPAD + k0 + k] = (_Float16)tile[k][v];
  }
}

// Zero the K-pad columns [100,128) of hs16 (written once; recurrence fills K<100)
__global__ __launch_bounds__(256) void zero_hs_pad(_Float16* __restrict__ hs16) {
  int idx = blockIdx.x * 256 + threadIdx.x;
  if (idx < NROWS * (KPAD - HID)) {
    int r = idx / (KPAD - HID);
    int c = HID + idx % (KPAD - HID);
    hs16[(size_t)r * KPAD + c] = (_Float16)0.f;
  }
}

// ---------------------------------------------------------------------------
// Kernel B: persistent single-workgroup recurrence using V_WMMA_F32_16X16X4_F32.
// h[16,100] @ W_hh[100,100]: M=16 (batch), K=100 (25 k-steps of 4), N=112 (7 tiles).
// Wave w (w<7) owns N-tile w; W_hh^T and h live in LDS.
// ---------------------------------------------------------------------------
__global__ __launch_bounds__(256) void rnn_recurrence(
    const float* __restrict__ pre,        // [S][B][H]
    const float* __restrict__ hidden0,    // [B][H]
    const float* __restrict__ W_hh,       // [H][H]
    _Float16* __restrict__ hs16,          // [4096][128]
    float* __restrict__ h_last) {         // [B][H]
  __shared__ __align__(16) float wT[112 * WT_STRIDE];   // wT[n][k] = W_hh[k][n]
  __shared__ __align__(16) float h[BATCH * H_STRIDE];   // h[b][k]
  const int tid = threadIdx.x;

  for (int i = tid; i < 112 * HID; i += 256) {
    int n = i / HID, k = i - n * HID;
    wT[n * WT_STRIDE + k] = (n < HID) ? W_hh[k * HID + n] : 0.0f;
  }
  for (int i = tid; i < BATCH * HID; i += 256) {
    int b = i / HID, k = i - b * HID;
    h[b * H_STRIDE + k] = hidden0[i];
  }
  __syncthreads();

  const int wv = tid >> 5;
  const int lane = tid & 31;
  const bool active = (wv < 7);
  const int nc = wv * 16 + (lane & 15);        // this lane's output column
  const int loff = (lane & 16) ? 2 : 0;        // K sub-offset per 16x4 f32 A/B layout
  const int brow = (lane & 16) ? 8 : 0;        // C-fragment row base for hi half-wave

  for (int t = 0; t < SEQ; ++t) {
    v8f c = {};
    if (active) {
#if __has_builtin(__builtin_amdgcn_wmma_f32_16x16x4_f32)
#pragma unroll 5
      for (int k4 = 0; k4 < 25; ++k4) {
        int kk = k4 * 4 + loff;
        v2f a = *(const v2f*)&h[(lane & 15) * H_STRIDE + kk];   // A: M=lane&15, K pair
        v2f bm = *(const v2f*)&wT[nc * WT_STRIDE + kk];         // B: N=nc, K pair
        c = __builtin_amdgcn_wmma_f32_16x16x4_f32(
            false, a, false, bm, (short)0, c, false, false);
      }
#else
#pragma unroll
      for (int j = 0; j < 8; ++j) {
        int b = j + brow;
        float acc = 0.f;
        for (int k = 0; k < HID; ++k)
          acc = fmaf(h[b * H_STRIDE + k], wT[nc * WT_STRIDE + k], acc);
        c[j] = acc;
      }
#endif
    }
    __syncthreads();   // all waves done reading h for step t
    if (active) {
      const float* pt = pre + t * (BATCH * HID);
#pragma unroll
      for (int j = 0; j < 8; ++j) {
        int b = j + brow;                       // C layout: VGPR j -> M=j (+8 hi lanes)
        if (nc < HID) {
          float val = tanhf(c[j] + pt[b * HID + nc]);
          h[b * H_STRIDE + nc] = val;
          hs16[(size_t)(b * SEQ + t) * KPAD + nc] = (_Float16)val;
          if (t == SEQ - 1) h_last[b * HID + nc] = val;
        }
      }
    }
    __syncthreads();   // h updates visible before step t+1
  }
}

// ---------------------------------------------------------------------------
// Kernel C1: logits = hs16[4096,128] @ wT^T -> [4096,32000] via
// V_WMMA_F32_16X16X32_F16. Block = 8 waves; each wave: 1 M-tile x 1 N-tile,
// K = 4 steps of 32. Per-lane A/B fragment = two b128 loads at +0/+32 bytes
// (rows padded to 256 B so the ISA 16-bit operand layout maps to row-major).
// ---------------------------------------------------------------------------
__global__ __launch_bounds__(256) void logits_gemm(
    const _Float16* __restrict__ hs16,   // [4096][128]
    const _Float16* __restrict__ wT,     // [32000][128]
    const float* __restrict__ b_fc,      // [32000]
    float* __restrict__ out) {           // [4096][32000]
  const int lane = threadIdx.x & 31;
  const int wv = threadIdx.x >> 5;
  const int m = blockIdx.y;                  // 0..255  M-tile
  const int ntile = blockIdx.x * 8 + wv;     // 0..1999 N-tile
  const int hi = (lane & 16) ? 8 : 0;        // half-wave K offset (halves)

  const _Float16* arow = hs16 + (size_t)(m * 16 + (lane & 15)) * KPAD + hi;
  const _Float16* bcol = wT + (size_t)(ntile * 16 + (lane & 15)) * KPAD + hi;

  v8f c = {};
#pragma unroll
  for (int kb = 0; kb < 4; ++kb) {
    v8h a0 = *(const v8h*)(arow + kb * 32);
    v8h a1 = *(const v8h*)(arow + kb * 32 + 16);
    v8h b0 = *(const v8h*)(bcol + kb * 32);
    v8h b1 = *(const v8h*)(bcol + kb * 32 + 16);
    v16h a = __builtin_shufflevector(a0, a1, 0, 1, 2, 3, 4, 5, 6, 7,
                                     8, 9, 10, 11, 12, 13, 14, 15);
    v16h b = __builtin_shufflevector(b0, b1, 0, 1, 2, 3, 4, 5, 6, 7,
                                     8, 9, 10, 11, 12, 13, 14, 15);
    c = __builtin_amdgcn_wmma_f32_16x16x32_f16(
        false, a, false, b, (short)0, c, false, false);
  }

  float bias = b_fc[ntile * 16 + (lane & 15)];        // lane's column is fixed
  size_t base = (size_t)(m * 16) * VOCAB + (size_t)(ntile * 16 + (lane & 15));
#pragma unroll
  for (int j = 0; j < 8; ++j) {
    int row = j + ((lane & 16) ? 8 : 0);
    out[base + (size_t)row * VOCAB] = c[j] + bias;
  }
}

// ---------------------------------------------------------------------------
// Kernel C2: in-place log_softmax over V=32000. One block per row; 32 values
// per thread stay in registers -> exactly one read + one write of d_out.
// ---------------------------------------------------------------------------
__global__ __launch_bounds__(1024) void logsoftmax_inplace(float* __restrict__ out) {
  __shared__ float red[32];
  const int tid = threadIdx.x;
  const int wid = tid >> 5, lane = tid & 31;
  float* row = out + (size_t)blockIdx.x * VOCAB;

  float v[32];
  float mx = -INFINITY;
#pragma unroll
  for (int i = 0; i < 32; ++i) {
    int idx = tid + i * 1024;
    v[i] = (idx < VOCAB) ? row[idx] : -INFINITY;
    mx = fmaxf(mx, v[i]);
  }
  for (int off = 16; off; off >>= 1) mx = fmaxf(mx, __shfl_xor(mx, off, 32));
  if (lane == 0) red[wid] = mx;
  __syncthreads();
  if (wid == 0) {
    float m2 = red[lane];
    for (int off = 16; off; off >>= 1) m2 = fmaxf(m2, __shfl_xor(m2, off, 32));
    if (lane == 0) red[0] = m2;
  }
  __syncthreads();
  mx = red[0];
  __syncthreads();

  float s = 0.f;
#pragma unroll
  for (int i = 0; i < 32; ++i) {
    int idx = tid + i * 1024;
    if (idx < VOCAB) s += expf(v[i] - mx);
  }
  for (int off = 16; off; off >>= 1) s += __shfl_xor(s, off, 32);
  if (lane == 0) red[wid] = s;
  __syncthreads();
  if (wid == 0) {
    float s2 = red[lane];
    for (int off = 16; off; off >>= 1) s2 += __shfl_xor(s2, off, 32);
    if (lane == 0) red[0] = s2;
  }
  __syncthreads();
  float lse = logf(red[0]);

#pragma unroll
  for (int i = 0; i < 32; ++i) {
    int idx = tid + i * 1024;
    if (idx < VOCAB) row[idx] = v[i] - mx - lse;
  }
}

// ---------------------------------------------------------------------------
extern "C" void kernel_launch(void* const* d_in, const int* in_sizes, int n_in,
                              void* d_out, int out_size, void* d_ws, size_t ws_size,
                              hipStream_t stream) {
  const int*   x    = (const int*)  d_in[0];
  const float* hid0 = (const float*)d_in[1];
  const float* emb  = (const float*)d_in[2];
  const float* W_ih = (const float*)d_in[3];
  const float* W_hh = (const float*)d_in[4];
  const float* b_ih = (const float*)d_in[5];
  const float* b_hh = (const float*)d_in[6];
  const float* W_fc = (const float*)d_in[7];
  const float* b_fc = (const float*)d_in[8];
  float* out = (float*)d_out;

  char* ws = (char*)d_ws;
  float*    pre  = (float*)(ws);                 // 1,638,400 B
  _Float16* hs16 = (_Float16*)(ws + 1638400);    // 1,048,576 B
  _Float16* wT   = (_Float16*)(ws + 2686976);    // 8,192,000 B

  pre_kernel<<<1600, 256, 0, stream>>>(x, emb, W_ih, b_ih, b_hh, pre);
  transpose_wfc<<<dim3(500, 8), 256, 0, stream>>>(W_fc, wT);
  zero_hs_pad<<<(NROWS * (KPAD - HID) + 255) / 256, 256, 0, stream>>>(hs16);
  rnn_recurrence<<<1, 256, 0, stream>>>(pre, hid0, W_hh, hs16,
                                        out + (size_t)NROWS * VOCAB);
  logits_gemm<<<dim3(250, 256), 256, 0, stream>>>(hs16, wT, b_fc, out);
  logsoftmax_inplace<<<4096, 1024, 0, stream>>>(out);
}
